// MultiHeadAttentionDecoder_274877907085
// MI455X (gfx1250) — matrix operand order repaired and checked by reading, not compiled, not run on hardware
//
#include <hip/hip_runtime.h>

typedef __attribute__((ext_vector_type(16))) _Float16 v16h;
typedef __attribute__((ext_vector_type(8)))  _Float16 v8h;
typedef __attribute__((ext_vector_type(8)))  float    v8f;
typedef __attribute__((ext_vector_type(4)))  unsigned int u32x4;
typedef __attribute__((ext_vector_type(8)))  int      i32x8;
typedef __attribute__((ext_vector_type(4)))  int      i32x4;

#define EMBED 1024
#define NHEAD 16
#define HS    64
#define TNEW  2048
#define TPAST 2048
#define TTOT  4096
#define QB    64
#define KC    128

#if __has_builtin(__builtin_amdgcn_tensor_load_to_lds)
#define HAVE_TDM 1
#endif

// ---- WMMA fragment helpers (layouts per CDNA5 ISA 7.12.2, wave32) ----
// A (16x32 f16): lane<16 -> M=lane, K halves {k0..k0+7, k0+16..k0+23}, k0=0;
//                lane>=16 -> same M=lane-16 pattern with k0=8.
__device__ __forceinline__ v16h load_frag_a(const _Float16* base, int ld, int lane) {
    const int m  = lane & 15;
    const int k0 = (lane >> 4) << 3;            // 0 or 8
    const _Float16* p = base + m * ld + k0;
    v8h lo = *(const v8h*)(p);
    v8h hi = *(const v8h*)(p + 16);
    return __builtin_shufflevector(lo, hi, 0,1,2,3,4,5,6,7,8,9,10,11,12,13,14,15);
}

// B (32x16 f16): lane<16 -> N=lane, K {0..7, 16..23}; lane>=16 -> N=lane-16, K {8..15, 24..31}.
// `base` points at element (k=0, n=0) of a layout where (k,n) lives at n*ld + k.
__device__ __forceinline__ v16h load_frag_b(const _Float16* base, int ld, int lane) {
    const int n  = lane & 15;
    const int k0 = (lane >> 4) << 3;            // 0 or 8
    const _Float16* p = base + n * ld + k0;
    v8h lo = *(const v8h*)(p);
    v8h hi = *(const v8h*)(p + 16);
    return __builtin_shufflevector(lo, hi, 0,1,2,3,4,5,6,7,8,9,10,11,12,13,14,15);
}

__device__ __forceinline__ v8f wmma_f16(v16h a, v16h b, v8f c) {
    return __builtin_amdgcn_wmma_f32_16x16x32_f16(false, a, false, b, (short)0, c, false, false);
}

// ---- prep kernels ----
__global__ void cvt_f32_f16(const float* __restrict__ src, _Float16* __restrict__ dst, int n) {
    int i = blockIdx.x * blockDim.x + threadIdx.x;
    if (i < n) dst[i] = (_Float16)src[i];
}

// copy past K/V into the f32 outputs (front T_past slots) + build f16 K and f16 V^T
__global__ void past_copy(const float* __restrict__ pk, const float* __restrict__ pv,
                          float* __restrict__ ko, float* __restrict__ vo,
                          _Float16* __restrict__ kh, _Float16* __restrict__ vt) {
    int i = blockIdx.x * blockDim.x + threadIdx.x;
    const int n = 2 * NHEAD * TPAST * HS;
    if (i >= n) return;
    int d  = i & (HS - 1);
    int t  = (i >> 6) & (TPAST - 1);
    int bh = i >> 17;
    float k = pk[i], v = pv[i];
    size_t dsti = ((size_t)bh * TTOT + t) * HS + d;
    ko[dsti] = k;  kh[dsti] = (_Float16)k;
    vo[dsti] = v;
    vt[((size_t)bh * HS + d) * TTOT + t] = (_Float16)v;
}

// ---- Q/K/V projection: y[t,e] = sum_d x[t,d] * W[e,d]  (per head, 16-row tiles) ----
__global__ void __launch_bounds__(128) qkv_proj(
    const _Float16* __restrict__ xh, const _Float16* __restrict__ wq,
    const _Float16* __restrict__ wk, const _Float16* __restrict__ wv,
    _Float16* __restrict__ qh, _Float16* __restrict__ kh, _Float16* __restrict__ vt,
    float* __restrict__ ko, float* __restrict__ vo) {
    const int lane  = threadIdx.x & 31;
    const int gw    = blockIdx.x * 4 + (threadIdx.x >> 5);
    const int ttile = gw & 127;                  // TNEW/16
    const int bh    = gw >> 7;                   // 0..31
    const int b     = bh >> 4, h = bh & 15;
    const int t0    = ttile << 4;

    const _Float16* xbase = xh + ((size_t)(b * TNEW + t0)) * EMBED + h * HS;
    v16h a0 = load_frag_a(xbase,      EMBED, lane);
    v16h a1 = load_frag_a(xbase + 32, EMBED, lane);

    const int col   = lane & 15;
    const int rbase = (lane >> 4) * 8;

    for (int p = 0; p < 3; ++p) {
        const _Float16* w = (p == 0) ? wq : ((p == 1) ? wk : wv);
        for (int et = 0; et < 4; ++et) {
            v8f acc = {};
            v16h b0 = load_frag_b(w + et * 16 * HS,      HS, lane);  // W[e,d], d contig
            v16h b1 = load_frag_b(w + et * 16 * HS + 32, HS, lane);
            acc = wmma_f16(a0, b0, acc);
            acc = wmma_f16(a1, b1, acc);
            if (p == 0) {
                _Float16* dst = qh + ((size_t)bh * TNEW + t0) * HS + et * 16 + col;
                for (int r = 0; r < 8; ++r) dst[(size_t)(rbase + r) * HS] = (_Float16)acc[r];
            } else if (p == 1) {
                size_t o = ((size_t)bh * TTOT + TPAST + t0) * HS + et * 16 + col;
                for (int r = 0; r < 8; ++r) {
                    float xv = acc[r];
                    ko[o + (size_t)(rbase + r) * HS] = xv;
                    kh[o + (size_t)(rbase + r) * HS] = (_Float16)xv;
                }
            } else {
                size_t o  = ((size_t)bh * TTOT + TPAST + t0) * HS + et * 16 + col;
                size_t ot = ((size_t)bh * HS + et * 16 + col) * TTOT + TPAST + t0;
                for (int r = 0; r < 8; ++r) {
                    float xv = acc[r];
                    vo[o + (size_t)(rbase + r) * HS] = xv;
                    vt[ot + rbase + r] = (_Float16)xv;
                }
            }
        }
    }
}

// ---- flash attention: 1 block = (b,h, 64 q rows); 4 waves x 16 rows each ----
// V chunk [64 x KC] staged to LDS by the Tensor Data Mover (one DMA per chunk,
// shared by all 4 waves; overlapped with QK^T WMMAs + softmax, drained with
// s_wait_tensorcnt before the P*V stage).
__global__ void __launch_bounds__(128) attn_kernel(
    const _Float16* __restrict__ qh, const _Float16* __restrict__ kh,
    const _Float16* __restrict__ vt, const unsigned char* __restrict__ pad,
    _Float16* __restrict__ aout) {
    __shared__ __align__(16) float    sS[QB][KC];   // scores; reused in-place as f16 P
    __shared__ __align__(32) _Float16 sV[HS][KC];   // V^T chunk (TDM destination)
    __shared__ float sM[QB], sL[QB], sA[QB];
    __shared__ unsigned char sPad[KC];

    const int tid   = threadIdx.x;
    const int lane  = tid & 31;
    const int w     = tid >> 5;
    const int qtile = blockIdx.x & 31;           // TNEW/QB
    const int bh    = blockIdx.x >> 5;
    const int b     = bh >> 4, h = bh & 15;
    const int qb    = qtile * QB;
    const float scale = 0.125f;                  // 1/sqrt(64)
    const float NEGINF = -__builtin_inff();

    const _Float16* qbase = qh + ((size_t)bh * TNEW + qb + w * 16) * HS;
    v16h aq0 = load_frag_a(qbase,      HS, lane);
    v16h aq1 = load_frag_a(qbase + 32, HS, lane);

    v8f o0 = {}, o1 = {}, o2 = {}, o3 = {};
    if (tid < QB) { sM[tid] = NEGINF; sL[tid] = 0.f; sA[tid] = 1.f; }

    const unsigned char* pm = pad + (size_t)b * TTOT;
    int nChunks = (qb + QB - 1 + TPAST) / KC + 1;     // causal early exit
    if (nChunks > TTOT / KC) nChunks = TTOT / KC;

    _Float16* sP = (_Float16*)&sS[0][0];
    const int PLD = KC * 2;                            // halves per P row

    const int col0  = lane & 15;
    const int rbase = (lane >> 4) * 8;

    for (int c = 0; c < nChunks; ++c) {
        const int key0 = c * KC;
        sPad[tid] = pm[key0 + tid];
        __syncthreads();                // A: prev-iter sP/sV reads done; sPad visible

        // --- kick off V-chunk DMA: global V^T [64 rows x KC halves, stride TTOT] -> LDS
#ifdef HAVE_TDM
        if (tid < 32) {                 // one wave issues; descriptor is WG-uniform (SGPRs)
            unsigned long long ga =
                (unsigned long long)(size_t)(vt + ((size_t)bh * HS) * TTOT + key0);
            unsigned ldsa = (unsigned)(size_t)&sV[0][0];
            u32x4 g0 = { 1u,                                   // count=1 (user D#)
                         ldsa,                                 // lds_addr
                         (unsigned)(ga & 0xFFFFFFFFu),         // global_addr[31:0]
                         (unsigned)((ga >> 32) & 0x01FFFFFFu) | (2u << 30) }; // [56:32] | type=2
            i32x8 g1 = { (int)(1u << 16),                      // mask=0, data_size=2B
                         (int)((unsigned)KC << 16),            // tensor_dim0[15:0]=KC
                         (int)((unsigned)HS << 16),            // dim0 hi=0, tensor_dim1=HS
                         (int)((unsigned)KC << 16),            // dim1 hi=0, tile_dim0=KC
                         HS,                                   // tile_dim1=HS, tile_dim2=0
                         TTOT,                                 // tensor_dim0_stride lo
                         0, 0 };                               // stride hi, dim1_stride
            i32x4 z4 = {};
#if defined(__clang_major__) && __clang_major__ >= 23
            i32x8 z8 = {};
            __builtin_amdgcn_tensor_load_to_lds(g0, g1, z4, z4, z8, 0);
#else
            __builtin_amdgcn_tensor_load_to_lds(g0, g1, z4, z4, 0);
#endif
        }
#else
        for (int i = tid * 8; i < HS * KC; i += 128 * 8) {     // cooperative fallback
            int d = i / KC, kk = i % KC;
            *(v8h*)&sV[d][kk] =
                *(const v8h*)(vt + ((size_t)bh * HS + d) * TTOT + key0 + kk);
        }
#endif

        // --- S = Q K^T (scaled, causally masked) -> LDS (overlaps the DMA)
        const _Float16* kbase = kh + ((size_t)bh * TTOT + key0) * HS;
        const int growbase = qb + w * 16 + rbase + TPAST;
        for (int kt = 0; kt < KC / 16; ++kt) {
            v8f acc = {};
            v16h b0 = load_frag_b(kbase + kt * 16 * HS,      HS, lane);
            v16h b1 = load_frag_b(kbase + kt * 16 * HS + 32, HS, lane);
            acc = wmma_f16(aq0, b0, acc);
            acc = wmma_f16(aq1, b1, acc);
            const int colg = kt * 16 + col0;
            const int gk = key0 + colg;
            const bool padok = sPad[colg] != 0;
            for (int r = 0; r < 8; ++r) {
                bool ok = padok && (gk <= growbase + r);
                sS[w * 16 + rbase + r][colg] = ok ? acc[r] * scale : NEGINF;
            }
        }
        __syncthreads();                // B

        // --- online softmax: one thread per q row; P written f16 in-place over sS
        if (tid < QB) {
            const int row = tid;
            float mo = sM[row], lo = sL[row];
            float cm = NEGINF;
            for (int cc = 0; cc < KC; ++cc) cm = fmaxf(cm, sS[row][cc]);
            float mn = fmaxf(mo, cm);
            float alpha, psum = 0.f;
            _Float16* prow = sP + row * PLD;
            if (mn == NEGINF) {                        // fully-masked so far
                alpha = 1.0f;
                for (int cc = 0; cc < KC; ++cc) prow[cc] = (_Float16)0.f;
            } else {
                alpha = __expf(mo - mn);               // mo = -inf -> 0
                for (int cc = 0; cc < KC; ++cc) {
                    float pv = __expf(sS[row][cc] - mn);
                    psum += pv;
                    prow[cc] = (_Float16)pv;           // in-place: write@2c < read@4c
                }
            }
            sM[row] = mn; sL[row] = lo * alpha + psum; sA[row] = alpha;
        }
#ifdef HAVE_TDM
        if (tid < 32) __builtin_amdgcn_s_wait_tensorcnt(0);   // drain DMA before release
#endif
        __syncthreads();                // C: sV + P + stats visible to all waves

        // --- rescale accumulators; O += P * V (B fragments from LDS-resident sV)
        float al[8];
        for (int r = 0; r < 8; ++r) al[r] = sA[w * 16 + rbase + r];
        for (int r = 0; r < 8; ++r) { o0[r] *= al[r]; o1[r] *= al[r]; o2[r] *= al[r]; o3[r] *= al[r]; }

        for (int kci = 0; kci < KC / 32; ++kci) {
            v16h ap = load_frag_a(sP + (w * 16) * PLD + kci * 32, PLD, lane);
            v16h b0 = load_frag_b(&sV[ 0][0] + kci * 32, KC, lane);
            v16h b1 = load_frag_b(&sV[16][0] + kci * 32, KC, lane);
            v16h b2 = load_frag_b(&sV[32][0] + kci * 32, KC, lane);
            v16h b3 = load_frag_b(&sV[48][0] + kci * 32, KC, lane);
            o0 = wmma_f16(ap, b0, o0);
            o1 = wmma_f16(ap, b1, o1);
            o2 = wmma_f16(ap, b2, o2);
            o3 = wmma_f16(ap, b3, o3);
        }
    }

    // normalize by l and emit f16 heads-interleaved [B,T,1024] for the out-proj GEMM
    for (int r = 0; r < 8; ++r) {
        const int row = w * 16 + rbase + r;
        float l = sL[row];
        float inv = (l > 0.f) ? 1.f / l : 0.f;
        size_t off = ((size_t)(b * TNEW + qb + row)) * EMBED + h * HS + col0;
        aout[off +  0] = (_Float16)(o0[r] * inv);
        aout[off + 16] = (_Float16)(o1[r] * inv);
        aout[off + 32] = (_Float16)(o2[r] * inv);
        aout[off + 48] = (_Float16)(o3[r] * inv);
    }
}

// ---- out = attn @ Wo^T + bo ----
__global__ void __launch_bounds__(128) out_proj(
    const _Float16* __restrict__ ah, const _Float16* __restrict__ woh,
    const float* __restrict__ bo, float* __restrict__ out) {
    const int lane = threadIdx.x & 31;
    const int gw   = blockIdx.x * 4 + (threadIdx.x >> 5);
    const int ot   = gw & 63;                    // EMBED/16 col tiles
    const int rt   = gw >> 6;                    // B*TNEW/16 row tiles
    const int col   = lane & 15;
    const int rbase = (lane >> 4) * 8;
    const float bias = bo[ot * 16 + col];
    v8f acc = {bias, bias, bias, bias, bias, bias, bias, bias};
    const _Float16* abase = ah  + (size_t)rt * 16 * EMBED;
    const _Float16* wbase = woh + (size_t)ot * 16 * EMBED;   // Wo[o,i], i contig = B layout
    for (int kci = 0; kci < EMBED / 32; ++kci) {
        v16h a = load_frag_a(abase + kci * 32, EMBED, lane);
        v16h b = load_frag_b(wbase + kci * 32, EMBED, lane);
        acc = wmma_f16(a, b, acc);
    }
    float* dst = out + (size_t)rt * 16 * EMBED + ot * 16 + col;
    for (int r = 0; r < 8; ++r) dst[(size_t)(rbase + r) * EMBED] = acc[r];
}

extern "C" void kernel_launch(void* const* d_in, const int* in_sizes, int n_in,
                              void* d_out, int out_size, void* d_ws, size_t ws_size,
                              hipStream_t stream) {
    const float* x          = (const float*)d_in[0];
    const unsigned char* pm = (const unsigned char*)d_in[1];
    const float* pk         = (const float*)d_in[2];
    const float* pv         = (const float*)d_in[3];
    const float* wq         = (const float*)d_in[4];
    const float* wk         = (const float*)d_in[5];
    const float* wv         = (const float*)d_in[6];
    const float* wo         = (const float*)d_in[7];
    const float* bo         = (const float*)d_in[8];

    float* out = (float*)d_out;
    float* ko  = out + (size_t)2 * TNEW * EMBED;
    float* vo  = ko  + (size_t)2 * NHEAD * TTOT * HS;

    char* ws = (char*)d_ws;
    size_t off = 0;
    auto alloc = [&](size_t bytes) -> char* {
        char* p = ws + off; off += (bytes + 255) & ~(size_t)255; return p;
    };
    _Float16* xh  = (_Float16*)alloc((size_t)2 * TNEW * EMBED * 2);   // 8 MB
    _Float16* qh  = (_Float16*)alloc((size_t)32 * TNEW * HS * 2);     // 8 MB
    _Float16* khh = (_Float16*)alloc((size_t)32 * TTOT * HS * 2);     // 16 MB
    _Float16* vth = (_Float16*)alloc((size_t)32 * HS * TTOT * 2);     // 16 MB (V^T)
    _Float16* ath = (_Float16*)alloc((size_t)2 * TNEW * EMBED * 2);   // 8 MB
    _Float16* wqh = (_Float16*)alloc((size_t)HS * HS * 2);
    _Float16* wkh = (_Float16*)alloc((size_t)HS * HS * 2);
    _Float16* wvh = (_Float16*)alloc((size_t)HS * HS * 2);
    _Float16* woh = (_Float16*)alloc((size_t)EMBED * EMBED * 2);      // 2 MB

    int n;
    n = 2 * TNEW * EMBED;     cvt_f32_f16<<<(n + 255) / 256, 256, 0, stream>>>(x,  xh,  n);
    n = HS * HS;              cvt_f32_f16<<<(n + 255) / 256, 256, 0, stream>>>(wq, wqh, n);
    n = HS * HS;              cvt_f32_f16<<<(n + 255) / 256, 256, 0, stream>>>(wk, wkh, n);
    n = HS * HS;              cvt_f32_f16<<<(n + 255) / 256, 256, 0, stream>>>(wv, wvh, n);
    n = EMBED * EMBED;        cvt_f32_f16<<<(n + 255) / 256, 256, 0, stream>>>(wo, woh, n);
    n = 2 * NHEAD * TPAST * HS;
    past_copy<<<(n + 255) / 256, 256, 0, stream>>>(pk, pv, ko, vo, khh, vth);

    qkv_proj   <<<1024, 128, 0, stream>>>(xh, wqh, wkh, wvh, qh, khh, vth, ko, vo);
    attn_kernel<<<1024, 128, 0, stream>>>(qh, khh, vth, pm, ath);
    out_proj   <<<4096, 128, 0, stream>>>(ath, woh, bo, out);
}
